// HyperGraph_77077483094367
// MI455X (gfx1250) — matrix operand order repaired and checked by reading, not compiled, not run on hardware
//
#include <hip/hip_runtime.h>
#include <math.h>

// ---------------------------------------------------------------------------
// HyperGraph attention pooling for MI455X (gfx1250, wave32, WMMA).
//
// Three GEMMs (126 GFLOP, K up to 20000) via v_wmma_f32_16x16x32_bf16
// (H is binary -> exact in bf16; f32 accumulation). Double-buffered LDS
// tiles, one barrier per K-step, guard-free steady-state loops (tail tile of
// GEMM2 handled by a separately-launched guarded block).
// Memory bound: ~1.2 GB HBM traffic ~= 50us @ 23.3 TB/s.
// ---------------------------------------------------------------------------

typedef __attribute__((ext_vector_type(16))) __bf16 v16bf;
typedef __attribute__((ext_vector_type(8)))  __bf16 v8bf;
typedef __attribute__((ext_vector_type(8)))  float  v8f;

#define EPSF 1e-12f

// ---------------------------------------------------------------------------
// Transpose + f32->bf16 convert: in [R,C] f32 -> out [R,C] bf16 (optional)
// and outT [C,R] bf16. R,C multiples of 32. Block (32,8).
// ---------------------------------------------------------------------------
__global__ __launch_bounds__(256)
void transpose_convert(const float* __restrict__ in, __bf16* __restrict__ out,
                       __bf16* __restrict__ outT, int R, int C) {
    __shared__ float tile[32][33];
    const int bx = blockIdx.x * 32;   // C offset
    const int by = blockIdx.y * 32;   // R offset
    const int tx = threadIdx.x;       // 0..31
    const int ty = threadIdx.y;       // 0..7
    #pragma unroll
    for (int i = 0; i < 32; i += 8) {
        int r = by + ty + i, c = bx + tx;
        float v = in[(size_t)r * C + c];
        tile[ty + i][tx] = v;
        if (out) out[(size_t)r * C + c] = (__bf16)v;
    }
    __syncthreads();
    #pragma unroll
    for (int i = 0; i < 32; i += 8) {
        int c = bx + ty + i, r = by + tx;
        outT[(size_t)c * R + r] = (__bf16)tile[tx][ty + i];
    }
}

// ---------------------------------------------------------------------------
// Row-sum reciprocal (1/clip(sum,EPS)): one block per row. Vectorized loads.
// ---------------------------------------------------------------------------
__global__ __launch_bounds__(256)
void rowsum_inv_f32(const float* __restrict__ in, float* __restrict__ out_inv, int C) {
    __shared__ float red[256];
    const int row = blockIdx.x, tid = threadIdx.x;
    const float4* p = reinterpret_cast<const float4*>(in + (size_t)row * C);
    float s = 0.f;
    for (int c = tid; c < (C >> 2); c += 256) {
        float4 v = p[c];
        s += v.x + v.y + v.z + v.w;
    }
    red[tid] = s; __syncthreads();
    for (int st = 128; st > 0; st >>= 1) {
        if (tid < st) red[tid] += red[tid + st];
        __syncthreads();
    }
    if (tid == 0) out_inv[row] = 1.0f / fmaxf(red[0], EPSF);
}

__global__ __launch_bounds__(256)
void rowsum_inv_bf16(const __bf16* __restrict__ in, float* __restrict__ out_inv, int C) {
    __shared__ float red[256];
    const int row = blockIdx.x, tid = threadIdx.x;
    const v8bf* p = reinterpret_cast<const v8bf*>(in + (size_t)row * C);
    float s = 0.f;
    for (int c = tid; c < (C >> 3); c += 256) {
        v8bf v = p[c];
        #pragma unroll
        for (int i = 0; i < 8; ++i) s += (float)v[i];
    }
    red[tid] = s; __syncthreads();
    for (int st = 128; st > 0; st >>= 1) {
        if (tid < st) red[tid] += red[tid + st];
        __syncthreads();
    }
    if (tid == 0) out_inv[row] = 1.0f / fmaxf(red[0], EPSF);
}

// ---------------------------------------------------------------------------
// Tile-load helpers (A: 128x32 bf16 as 512 x 16B segments; B: 64x32 as 256).
// ---------------------------------------------------------------------------
template <bool GUARD>
__device__ __forceinline__ v8bf load_a_seg(const __bf16* __restrict__ A, int lda,
                                           int m0, int M, int k0, int seg,
                                           const v8bf& zerov) {
    const int row = seg >> 2, part = seg & 3;
    const int gm = m0 + row;
    if (!GUARD || gm < M)
        return *reinterpret_cast<const v8bf*>(A + (size_t)gm * lda + k0 + part * 8);
    return zerov;
}

__device__ __forceinline__ v8bf load_b_seg(const __bf16* __restrict__ BT, int ldb,
                                           int n0, int k0, int seg) {
    const int col = seg >> 2, part = seg & 3;
    return *reinterpret_cast<const v8bf*>(BT + (size_t)(n0 + col) * ldb + k0 + part * 8);
}

// ---------------------------------------------------------------------------
// WMMA GEMM: C[M,D] = rowscale[m] * (A[M,K] * B[K,D]),  B given transposed
// (BT[D,K] row-major so B columns are per-lane K-contiguous).  Optional
// fusion v = 0.5*(Xfuse[m,n] + v).  Writes f32 Cout [M,D] and/or bf16
// CTout [D,ctld].  Block tile 128(M) x 64(N), K-step 32, double-buffered LDS,
// 8 wave32 waves each computing 32x32 via 2x2 v_wmma_f32_16x16x32_bf16.
// K % 32 == 0, D % 64 == 0.  GUARD=false requires mbase+128 <= M.
// ---------------------------------------------------------------------------
template <bool GUARD>
__global__ __launch_bounds__(256)
void gemm_bf16_wmma(const __bf16* __restrict__ A, const __bf16* __restrict__ BT,
                    const float* __restrict__ rowscale, const float* __restrict__ Xfuse,
                    float* __restrict__ Cout, __bf16* __restrict__ CTout,
                    int M, int K, int lda, int ldb, int ctld, int D, int mbase) {
    __shared__ __align__(16) __bf16 sA[2][128 * 32];
    __shared__ __align__(16) __bf16 sB[2][64 * 32];
    __shared__ float sC[128 * 65];   // padded: conflict-free in both epilogue passes

    const int tid  = threadIdx.x;
    const int lane = tid & 31;
    const int wave = tid >> 5;
    const int wm   = wave & 3;        // 4 waves along M (32 rows each)
    const int wn   = wave >> 2;       // 2 waves along N (32 cols each)
    const int m0   = mbase + blockIdx.y * 128;
    const int n0   = blockIdx.x * 64;

    const int  lmod = lane & 15;
    const bool low  = lane < 16;
    const int  ac0  = low ? 0 : 8;    // A frag: K chunks {ac0..+7, ac0+16..+23}
    const int  bk0  = low ? 0 : 8;    // B frag: K chunks {bk0..+7, bk0+16..+23}

    v8f acc[2][2];
    #pragma unroll
    for (int mi = 0; mi < 2; ++mi)
        #pragma unroll
        for (int ni = 0; ni < 2; ++ni)
            #pragma unroll
            for (int i = 0; i < 8; ++i) acc[mi][ni][i] = 0.0f;

    v8bf zerov;
    #pragma unroll
    for (int i = 0; i < 8; ++i) zerov[i] = (__bf16)0.0f;

    const int nk = K >> 5;

    // Prologue: tile 0 -> buffer 0.
    {
        v8bf a0 = load_a_seg<GUARD>(A, lda, m0, M, 0, tid, zerov);
        v8bf a1 = load_a_seg<GUARD>(A, lda, m0, M, 0, tid + 256, zerov);
        v8bf b0 = load_b_seg(BT, ldb, n0, 0, tid);
        {
            int row = tid >> 2, part = tid & 3;
            *reinterpret_cast<v8bf*>(&sA[0][row * 32 + part * 8]) = a0;
            int row1 = (tid + 256) >> 2, part1 = tid & 3;
            *reinterpret_cast<v8bf*>(&sA[0][row1 * 32 + part1 * 8]) = a1;
            int col = tid >> 2;
            *reinterpret_cast<v8bf*>(&sB[0][col * 32 + part * 8]) = b0;
        }
    }
    __syncthreads();

    for (int it = 0; it < nk; ++it) {
        const int  cur = it & 1;
        const int  nxtk = (it + 1) << 5;
        const bool hn = (it + 1 < nk);

        // Prefetch next tile into registers (overlaps the WMMAs below).
        v8bf pa0, pa1, pb;
        if (hn) {
            pa0 = load_a_seg<GUARD>(A, lda, m0, M, nxtk, tid, zerov);
            pa1 = load_a_seg<GUARD>(A, lda, m0, M, nxtk, tid + 256, zerov);
            pb  = load_b_seg(BT, ldb, n0, nxtk, tid);
        }

        // Fragment packing per CDNA5 ISA 7.12.2 (16-bit A 16x32, B 32x16).
        v16bf af[2], bfr[2];
        #pragma unroll
        for (int mi = 0; mi < 2; ++mi) {
            int row = wm * 32 + mi * 16 + lmod;
            v8bf lo = *reinterpret_cast<const v8bf*>(&sA[cur][row * 32 + ac0]);
            v8bf hi = *reinterpret_cast<const v8bf*>(&sA[cur][row * 32 + ac0 + 16]);
            #pragma unroll
            for (int i = 0; i < 8; ++i) { af[mi][i] = lo[i]; af[mi][i + 8] = hi[i]; }
        }
        #pragma unroll
        for (int ni = 0; ni < 2; ++ni) {
            int col = wn * 32 + ni * 16 + lmod;
            v8bf lo = *reinterpret_cast<const v8bf*>(&sB[cur][col * 32 + bk0]);
            v8bf hi = *reinterpret_cast<const v8bf*>(&sB[cur][col * 32 + bk0 + 16]);
            #pragma unroll
            for (int i = 0; i < 8; ++i) { bfr[ni][i] = lo[i]; bfr[ni][i + 8] = hi[i]; }
        }
        #pragma unroll
        for (int mi = 0; mi < 2; ++mi)
            #pragma unroll
            for (int ni = 0; ni < 2; ++ni)
                acc[mi][ni] = __builtin_amdgcn_wmma_f32_16x16x32_bf16(
                    false, af[mi], false, bfr[ni], (short)0, acc[mi][ni], false, false);

        // Commit prefetched tile into the other buffer.
        if (hn) {
            const int nb = cur ^ 1;
            int row = tid >> 2, part = tid & 3;
            *reinterpret_cast<v8bf*>(&sA[nb][row * 32 + part * 8]) = pa0;
            int row1 = (tid + 256) >> 2;
            *reinterpret_cast<v8bf*>(&sA[nb][row1 * 32 + part * 8]) = pa1;
            int col = tid >> 2;
            *reinterpret_cast<v8bf*>(&sB[nb][col * 32 + part * 8]) = pb;
        }
        __syncthreads();
    }

    // Stage C tile to LDS (C layout: M = vgpr + 8*(lane>=16), N = lane%16).
    const int mh = low ? 0 : 8;
    #pragma unroll
    for (int mi = 0; mi < 2; ++mi)
        #pragma unroll
        for (int ni = 0; ni < 2; ++ni) {
            int nl = wn * 32 + ni * 16 + lmod;
            #pragma unroll
            for (int i = 0; i < 8; ++i) {
                int ml = wm * 32 + mi * 16 + mh + i;
                sC[ml * 65 + nl] = acc[mi][ni][i];
            }
        }
    __syncthreads();

    if (Cout) {   // row-major f32 output, coalesced over n
        for (int idx = tid; idx < 128 * 64; idx += 256) {
            int ml = idx >> 6, nl = idx & 63;
            int m = m0 + ml, n = n0 + nl;
            if (!GUARD || m < M) {
                float v = sC[ml * 65 + nl] * rowscale[m];
                if (Xfuse) v = 0.5f * (Xfuse[(size_t)m * D + n] + v);
                Cout[(size_t)m * D + n] = v;
            }
        }
    }
    if (CTout) {  // transposed bf16 output, coalesced over m
        for (int idx = tid; idx < 128 * 64; idx += 256) {
            int nl = idx >> 7, ml = idx & 127;
            int m = m0 + ml, n = n0 + nl;
            if (!GUARD || m < M) {
                float v = sC[ml * 65 + nl] * rowscale[m];
                if (Xfuse) v = 0.5f * (Xfuse[(size_t)m * D + n] + v);
                CTout[(size_t)n * ctld + m] = (__bf16)v;
            }
        }
    }
}

// ---------------------------------------------------------------------------
// w = W_att @ q_att   (twoD x A  ·  A)
// ---------------------------------------------------------------------------
__global__ __launch_bounds__(256)
void wvec_kernel(const float* __restrict__ W, const float* __restrict__ q,
                 float* __restrict__ w, int twoD, int A) {
    int j = blockIdx.x * blockDim.x + threadIdx.x;
    if (j < twoD) {
        float s = 0.f;
        for (int a = 0; a < A; ++a) s += W[(size_t)j * A + a] * q[a];
        w[j] = s;
    }
}

// scores[e] = dot(codes[e,:], w[0:D]) + dot(edge_hid[e,:], w[D:2D]); one wave/edge
__global__ __launch_bounds__(256)
void scores_kernel(const float* __restrict__ codes, const float* __restrict__ eh,
                   const float* __restrict__ w, float* __restrict__ scores, int E, int D) {
    const int lane = threadIdx.x & 31;
    const int wave = threadIdx.x >> 5;
    const int e = blockIdx.x * 8 + wave;
    if (e >= E) return;
    float s = 0.f;
    for (int i = lane; i < D; i += 32)
        s += codes[(size_t)e * D + i] * w[i] + eh[(size_t)e * D + i] * w[D + i];
    for (int off = 16; off > 0; off >>= 1) s += __shfl_down(s, off, 32);
    if (lane == 0) scores[e] = s;
}

__global__ __launch_bounds__(1024)
void softmax_kernel(const float* __restrict__ scores, float* __restrict__ alpha, int E) {
    __shared__ float red[1024];
    const int tid = threadIdx.x;
    float mx = -3.4e38f;
    for (int e = tid; e < E; e += 1024) mx = fmaxf(mx, scores[e]);
    red[tid] = mx; __syncthreads();
    for (int s = 512; s > 0; s >>= 1) {
        if (tid < s) red[tid] = fmaxf(red[tid], red[tid + s]);
        __syncthreads();
    }
    mx = red[0]; __syncthreads();
    float sum = 0.f;
    for (int e = tid; e < E; e += 1024) sum += expf(scores[e] - mx);
    red[tid] = sum; __syncthreads();
    for (int s = 512; s > 0; s >>= 1) {
        if (tid < s) red[tid] += red[tid + s];
        __syncthreads();
    }
    float inv = 1.0f / red[0];
    for (int e = tid; e < E; e += 1024) alpha[e] = expf(scores[e] - mx) * inv;
}

// out[j] = sum_e alpha[e] * result[e, j], result = [codes | edge_hid]
__global__ __launch_bounds__(256)
void pool_kernel(const float* __restrict__ codes, const float* __restrict__ eh,
                 const float* __restrict__ alpha, float* __restrict__ out, int E, int D) {
    int j = blockIdx.x * blockDim.x + threadIdx.x;
    if (j >= 2 * D) return;
    const float* src = (j < D) ? codes : eh;
    int col = (j < D) ? j : j - D;
    float s = 0.f;
    for (int e = 0; e < E; ++e) s += alpha[e] * src[(size_t)e * D + col];
    out[j] = s;
}

// ---------------------------------------------------------------------------
extern "C" void kernel_launch(void* const* d_in, const int* in_sizes, int n_in,
                              void* d_out, int out_size, void* d_ws, size_t ws_size,
                              hipStream_t stream) {
    const float* X  = (const float*)d_in[0];   // [N, D]
    const float* H  = (const float*)d_in[1];   // [N, E]
    const float* Wt = (const float*)d_in[2];   // [2D, A]
    const float* q  = (const float*)d_in[3];   // [A]

    const int Aatt = in_sizes[3];
    const int twoD = in_sizes[2] / Aatt;
    const int D    = twoD / 2;
    const int N    = in_sizes[0] / D;
    const int E    = in_sizes[1] / N;
    (void)n_in; (void)out_size; (void)ws_size;

    char* ws = (char*)d_ws;
    size_t off = 0;
    auto take = [&](size_t bytes) -> void* {
        void* p = ws + off;
        off += (bytes + 255) & ~(size_t)255;
        return p;
    };
    __bf16* Hbf      = (__bf16*)take((size_t)N * E * 2);   // H  bf16 [N,E]
    __bf16* HTbf     = (__bf16*)take((size_t)E * N * 2);   // H^T bf16 [E,N]
    __bf16* XT       = (__bf16*)take((size_t)D * N * 2);   // X^T bf16 [D,N]
    __bf16* ehT      = (__bf16*)take((size_t)D * E * 2);   // edge_hid^T bf16 [D,E]
    __bf16* nmT      = (__bf16*)take((size_t)D * N * 2);   // node_msg^T bf16 [D,N]
    float*  edge_hid = (float*)take((size_t)E * D * 4);
    float*  codes    = (float*)take((size_t)E * D * 4);
    float*  dv_inv   = (float*)take((size_t)N * 4);
    float*  de_inv   = (float*)take((size_t)E * 4);
    float*  wvec     = (float*)take((size_t)twoD * 4);
    float*  scores   = (float*)take((size_t)E * 4);
    float*  alpha    = (float*)take((size_t)E * 4);

    dim3 tb(32, 8);
    // Convert H -> bf16 (+ transpose), X -> X^T bf16.
    transpose_convert<<<dim3(E / 32, N / 32), tb, 0, stream>>>(H, Hbf, HTbf, N, E);
    transpose_convert<<<dim3(D / 32, N / 32), tb, 0, stream>>>(X, (__bf16*)nullptr, XT, N, D);
    // Degrees.
    rowsum_inv_f32<<<N, 256, 0, stream>>>(H, dv_inv, E);
    rowsum_inv_bf16<<<E, 256, 0, stream>>>(HTbf, de_inv, N);

    const int gx = D / 64;
    // GEMM1: edge_hid = (H^T X) * de_inv   -> f32 [E,D] + bf16 transposed [D,E]
    gemm_bf16_wmma<false><<<dim3(gx, E / 128), 256, 0, stream>>>(
        HTbf, XT, de_inv, nullptr, edge_hid, ehT, E, N, N, N, E, D, 0);
    // GEMM2: node_msg = 0.5*(X + (H edge_hid) * dv_inv) -> bf16 transposed [D,N]
    {
        const int mfull = (N / 128) * 128;
        if (mfull > 0)
            gemm_bf16_wmma<false><<<dim3(gx, mfull / 128), 256, 0, stream>>>(
                Hbf, ehT, dv_inv, X, nullptr, nmT, N, E, E, E, N, D, 0);
        if (mfull < N)
            gemm_bf16_wmma<true><<<dim3(gx, 1), 256, 0, stream>>>(
                Hbf, ehT, dv_inv, X, nullptr, nmT, N, E, E, E, N, D, mfull);
    }
    // GEMM3: codes = (H^T node_msg) * de_inv -> f32 [E,D]
    gemm_bf16_wmma<false><<<dim3(gx, E / 128), 256, 0, stream>>>(
        HTbf, nmT, de_inv, nullptr, codes, nullptr, E, N, N, N, 0, D, 0);

    // Attention pooling tail.
    wvec_kernel<<<(twoD + 255) / 256, 256, 0, stream>>>(Wt, q, wvec, twoD, Aatt);
    scores_kernel<<<(E + 7) / 8, 256, 0, stream>>>(codes, edge_hid, wvec, scores, E, D);
    softmax_kernel<<<1, 1024, 0, stream>>>(scores, alpha, E);
    pool_kernel<<<(twoD + 255) / 256, 256, 0, stream>>>(codes, edge_hid, alpha,
                                                        (float*)d_out, E, D);
}
